// Cross_camera_21612275433689
// MI455X (gfx1250) — matrix operand order repaired and checked by reading, not compiled, not run on hardware
//
#include <hip/hip_runtime.h>

// ---------------- problem constants ----------------
constexpr int NC    = 6;      // cameras
constexpr int NID   = 751;    // ids
constexpr int NPAD  = 768;    // ids padded to 24*32 for WMMA tiles
constexpr int DIM   = 2048;   // feature dim
constexpr int NB    = 512;    // batch
constexpr int NPAIR = NC * NC;
constexpr int WARM  = 10;
constexpr int LW    = NC * NPAD;   // padded logits row width = 4608

// ---------------- WMMA types / helpers ----------------
typedef __attribute__((ext_vector_type(16))) __bf16 v16bf;
typedef __attribute__((ext_vector_type(8)))  __bf16 v8bf;
typedef __attribute__((ext_vector_type(8)))  float  v8f;

__device__ __forceinline__ v8f wmma_bf16(v16bf a, v16bf b, v8f c) {
  // D(f32 16x16) = A(bf16 16x32) * B(bf16 32x16) + C
  return __builtin_amdgcn_wmma_f32_16x16x32_bf16(
      /*neg_a=*/false, a, /*neg_b=*/false, b,
      /*c_mod=*/(short)0, c, /*reuse_a=*/false, /*reuse_b=*/false);
}

// A-frag (16x32 bf16): lane holds two 8-elem chunks: K = h*8 + [0..7] and 16 + h*8 + [0..7]
__device__ __forceinline__ v16bf load_afrag(const __bf16* p) {
  union U { v16bf v; v8bf h[2]; } u;
  u.h[0] = *(const v8bf*)(p);
  u.h[1] = *(const v8bf*)(p + 16);
  return u.v;
}

__device__ __forceinline__ v8f zero8() {
  v8f z;
#pragma unroll
  for (int i = 0; i < 8; ++i) z[i] = 0.f;
  return z;
}

__device__ __forceinline__ __bf16 f2bf(float f) {
  unsigned int u = __float_as_uint(f);
  unsigned int r = (u + 0x7FFFu + ((u >> 16) & 1u)) >> 16;
  unsigned short s = (unsigned short)r;
  return __builtin_bit_cast(__bf16, s);
}

// packed argmax key: monotone float key in high 32, (0x7FFFFFFF - idx) low.
// atomicMax -> max value wins; ties -> smallest index wins (matches jnp.argmin-first).
__device__ __forceinline__ unsigned long long packKI(float v, int idx) {
  unsigned int b = __float_as_uint(v);
  b = (b & 0x80000000u) ? ~b : (b | 0x80000000u);
  return ((unsigned long long)b << 32) | (unsigned int)(0x7FFFFFFF - idx);
}
__device__ __forceinline__ float unpackVal(unsigned long long pk) {
  unsigned int s = (unsigned int)(pk >> 32);
  s = (s & 0x80000000u) ? (s ^ 0x80000000u) : ~s;
  return __uint_as_float(s);
}
__device__ __forceinline__ int unpackIdx(unsigned long long pk) {
  return 0x7FFFFFFF - (int)(unsigned int)(pk & 0xFFFFFFFFu);
}

// ---------------- kernels ----------------

__global__ void k_zero64(unsigned long long* p, int n) {
  int i = blockIdx.x * 256 + threadIdx.x;
  if (i < n) p[i] = 0ULL;
}

// normalize features -> f32 copy + bf16 copy
__global__ void k_norm_feat(const float* __restrict__ feat,
                            float* __restrict__ fnorm, __bf16* __restrict__ fbf) {
  int b = blockIdx.x, t = threadIdx.x;
  const float* row = feat + (size_t)b * DIM;
  float ss = 0.f;
  for (int i = t; i < DIM; i += 256) { float v = row[i]; ss += v * v; }
  __shared__ float red[256];
  red[t] = ss; __syncthreads();
  for (int s = 128; s > 0; s >>= 1) { if (t < s) red[t] += red[t + s]; __syncthreads(); }
  float scale = 1.f / (sqrtf(red[0]) + 1e-12f);
  for (int i = t; i < DIM; i += 256) {
    float v = row[i] * scale;
    fnorm[(size_t)b * DIM + i] = v;
    fbf[(size_t)b * DIM + i]   = f2bf(v);
  }
}

// normalize cross_anchors -> padded bf16 [NC][NPAD][DIM] (pad rows = 0)
__global__ void k_anchors_bf16(const float* __restrict__ cross, __bf16* __restrict__ Abf) {
  int r = blockIdx.x, t = threadIdx.x;      // r in [0, NC*NPAD)
  int n = r % NPAD;
  __bf16* orow = Abf + (size_t)r * DIM;
  if (n >= NID) { for (int i = t; i < DIM; i += 256) orow[i] = f2bf(0.f); return; }
  int cam = r / NPAD;
  const float* row = cross + ((size_t)cam * NID + n) * DIM;
  float ss = 0.f;
  for (int i = t; i < DIM; i += 256) { float v = row[i]; ss += v * v; }
  __shared__ float red[256];
  red[t] = ss; __syncthreads();
  for (int s = 128; s > 0; s >>= 1) { if (t < s) red[t] += red[t + s]; __syncthreads(); }
  float scale = 1.f / (sqrtf(red[0]) + 1e-12f);
  for (int i = t; i < DIM; i += 256) orow[i] = f2bf(row[i] * scale);
}

// write x = normalize(cross)  (or normalize(intra) in warm-up)  to output
__global__ void k_write_x(const float* __restrict__ cross, const float* __restrict__ intra,
                          const int* __restrict__ epoch, float* __restrict__ x) {
  int r = blockIdx.x, t = threadIdx.x;      // r in [0, NC*NID)
  const float* src = (epoch[0] <= WARM) ? intra : cross;
  const float* row = src + (size_t)r * DIM;
  float ss = 0.f;
  for (int i = t; i < DIM; i += 256) { float v = row[i]; ss += v * v; }
  __shared__ float red[256];
  red[t] = ss; __syncthreads();
  for (int s = 128; s > 0; s >>= 1) { if (t < s) red[t] += red[t + s]; __syncthreads(); }
  float scale = 1.f / (sqrtf(red[0]) + 1e-12f);
  for (int i = t; i < DIM; i += 256) x[(size_t)r * DIM + i] = row[i] * scale;
}

// S tiles per camera pair + row/col argmax (rank / back) via packed atomic max.
// Each wave: 64x32 output (4x2 subtiles) -> 12 b128 loads per 8 WMMA.
__global__ void __launch_bounds__(256) k_pairsim(const __bf16* __restrict__ Abf,
                                                 unsigned long long* __restrict__ rowbest,
                                                 unsigned long long* __restrict__ colbest) {
  const int p = blockIdx.x;                  // pair = i*NC + c
  const int ci = p / NC, cc = p % NC;
  const int row0 = blockIdx.y * 64;
  const int wave = threadIdx.x >> 5, lane = threadIdx.x & 31;
  const __bf16* Ai = Abf + (size_t)ci * NPAD * DIM;
  const __bf16* Ac = Abf + (size_t)cc * NPAD * DIM;

  __shared__ float tile[8][64][33];
  __shared__ unsigned long long rowb[64];
  if (threadIdx.x < 64) rowb[threadIdx.x] = 0ULL;
  __syncthreads();

  const int lrow  = lane & 15;
  const int ahalf = (lane >> 4) << 3;   // 0 or 8
  const int bhalf = (lane >> 4) << 3;   // 0 or 8
  const __bf16* a0p = Ai + (size_t)(row0 + lrow) * DIM + ahalf;
  const __bf16* a1p = a0p + (size_t)16 * DIM;
  const __bf16* a2p = a0p + (size_t)32 * DIM;
  const __bf16* a3p = a0p + (size_t)48 * DIM;

  for (int cs = wave; cs < NPAD / 32; cs += 8) {
    const int col0 = cs * 32;
    const __bf16* b0p = Ac + (size_t)(col0 + lrow) * DIM + bhalf;
    const __bf16* b1p = b0p + (size_t)16 * DIM;
    v8f c00 = zero8(), c01 = zero8(), c10 = zero8(), c11 = zero8();
    v8f c20 = zero8(), c21 = zero8(), c30 = zero8(), c31 = zero8();
    for (int kk = 0; kk < DIM; kk += 32) {
      v16bf A0 = load_afrag(a0p + kk);
      v16bf A1 = load_afrag(a1p + kk);
      v16bf A2 = load_afrag(a2p + kk);
      v16bf A3 = load_afrag(a3p + kk);
      v16bf B0 = load_afrag(b0p + kk);
      v16bf B1 = load_afrag(b1p + kk);
      c00 = wmma_bf16(A0, B0, c00);  c01 = wmma_bf16(A0, B1, c01);
      c10 = wmma_bf16(A1, B0, c10);  c11 = wmma_bf16(A1, B1, c11);
      c20 = wmma_bf16(A2, B0, c20);  c21 = wmma_bf16(A2, B1, c21);
      c30 = wmma_bf16(A3, B0, c30);  c31 = wmma_bf16(A3, B1, c31);
    }
    const int crow = (lane >> 4) * 8, ccol = lane & 15;
#pragma unroll
    for (int r = 0; r < 8; ++r) {
      tile[wave][crow + r     ][ccol     ] = c00[r];
      tile[wave][crow + r     ][ccol + 16] = c01[r];
      tile[wave][crow + r + 16][ccol     ] = c10[r];
      tile[wave][crow + r + 16][ccol + 16] = c11[r];
      tile[wave][crow + r + 32][ccol     ] = c20[r];
      tile[wave][crow + r + 32][ccol + 16] = c21[r];
      tile[wave][crow + r + 48][ccol     ] = c30[r];
      tile[wave][crow + r + 48][ccol + 16] = c31[r];
    }
    // per-column best over valid rows of this strip -> global colbest (back)
    int rlim = NID - row0; if (rlim > 64) rlim = 64;
    int gc = col0 + lane;
    if (gc < NID) {
      float bv = -1e30f; int bi = 0;
      for (int r = 0; r < rlim; ++r) { float v = tile[wave][r][lane]; if (v > bv) { bv = v; bi = r; } }
      atomicMax(&colbest[(size_t)p * NPAD + gc], packKI(bv, row0 + bi));
    }
    // per-row best over valid cols of this strip -> LDS rowb (rank); 2 rows per lane
    int clim = NID - col0; if (clim > 32) clim = 32;
#pragma unroll
    for (int h = 0; h < 2; ++h) {
      int lr = lane + h * 32;
      if (row0 + lr < NID) {
        float bv = -1e30f; int bi = 0;
        for (int m = 0; m < clim; ++m) { float v = tile[wave][lr][m]; if (v > bv) { bv = v; bi = m; } }
        atomicMax(&rowb[lr], packKI(bv, col0 + bi));
      }
    }
  }
  __syncthreads();
  if (threadIdx.x < 64 && row0 + threadIdx.x < NID)
    rowbest[(size_t)p * NPAD + row0 + threadIdx.x] = rowb[threadIdx.x];
}

// logits = f_bf16 (512 x 2048)  x  A_cam^T  -> f32 [NB][NC*NPAD]; 64x32 per wave
__global__ void __launch_bounds__(256) k_logits(const __bf16* __restrict__ fbf,
                                                const __bf16* __restrict__ Abf,
                                                float* __restrict__ logits) {
  const int cam = blockIdx.x;
  const int row0 = blockIdx.y * 64;          // feature rows, 8 strips
  const int wave = threadIdx.x >> 5, lane = threadIdx.x & 31;
  const __bf16* Ac = Abf + (size_t)cam * NPAD * DIM;

  const int lrow  = lane & 15;
  const int ahalf = (lane >> 4) << 3;
  const int bhalf = (lane >> 4) << 3;
  const __bf16* a0p = fbf + (size_t)(row0 + lrow) * DIM + ahalf;
  const __bf16* a1p = a0p + (size_t)16 * DIM;
  const __bf16* a2p = a0p + (size_t)32 * DIM;
  const __bf16* a3p = a0p + (size_t)48 * DIM;

  for (int cs = wave; cs < NPAD / 32; cs += 8) {
    const int col0 = cs * 32;
    const __bf16* b0p = Ac + (size_t)(col0 + lrow) * DIM + bhalf;
    const __bf16* b1p = b0p + (size_t)16 * DIM;
    v8f c00 = zero8(), c01 = zero8(), c10 = zero8(), c11 = zero8();
    v8f c20 = zero8(), c21 = zero8(), c30 = zero8(), c31 = zero8();
    for (int kk = 0; kk < DIM; kk += 32) {
      v16bf A0 = load_afrag(a0p + kk);
      v16bf A1 = load_afrag(a1p + kk);
      v16bf A2 = load_afrag(a2p + kk);
      v16bf A3 = load_afrag(a3p + kk);
      v16bf B0 = load_afrag(b0p + kk);
      v16bf B1 = load_afrag(b1p + kk);
      c00 = wmma_bf16(A0, B0, c00);  c01 = wmma_bf16(A0, B1, c01);
      c10 = wmma_bf16(A1, B0, c10);  c11 = wmma_bf16(A1, B1, c11);
      c20 = wmma_bf16(A2, B0, c20);  c21 = wmma_bf16(A2, B1, c21);
      c30 = wmma_bf16(A3, B0, c30);  c31 = wmma_bf16(A3, B1, c31);
    }
    const int crow = (lane >> 4) * 8, ccol = lane & 15;
    const size_t cbase = (size_t)cam * NPAD + col0 + ccol;
#pragma unroll
    for (int r = 0; r < 8; ++r) {
      logits[(size_t)(row0 + crow + r     ) * LW + cbase     ] = c00[r];
      logits[(size_t)(row0 + crow + r     ) * LW + cbase + 16] = c01[r];
      logits[(size_t)(row0 + crow + r + 16) * LW + cbase     ] = c10[r];
      logits[(size_t)(row0 + crow + r + 16) * LW + cbase + 16] = c11[r];
      logits[(size_t)(row0 + crow + r + 32) * LW + cbase     ] = c20[r];
      logits[(size_t)(row0 + crow + r + 32) * LW + cbase + 16] = c21[r];
      logits[(size_t)(row0 + crow + r + 48) * LW + cbase     ] = c30[r];
      logits[(size_t)(row0 + crow + r + 48) * LW + cbase + 16] = c31[r];
    }
  }
}

// logsumexp over the 751 valid logits of each (b, cam) row
__global__ void k_lse(const float* __restrict__ logits, float* __restrict__ logZ) {
  int idx = blockIdx.x;                      // b*NC + cam
  int b = idx / NC, cam = idx % NC, t = threadIdx.x;
  const float* row = logits + (size_t)b * LW + (size_t)cam * NPAD;
  __shared__ float red[128];
  float mx = -1e30f;
  for (int i = t; i < NID; i += 128) mx = fmaxf(mx, row[i]);
  red[t] = mx; __syncthreads();
  for (int s = 64; s > 0; s >>= 1) { if (t < s) red[t] = fmaxf(red[t], red[t + s]); __syncthreads(); }
  mx = red[0]; __syncthreads();
  float sm = 0.f;
  for (int i = t; i < NID; i += 128) sm += expf(row[i] - mx);
  red[t] = sm; __syncthreads();
  for (int s = 64; s > 0; s >>= 1) { if (t < s) red[t] += red[t + s]; __syncthreads(); }
  if (t == 0) logZ[idx] = mx + logf(red[0]);
}

// per-batch gather: ce_self/ce_cross/validity -> scatter coefficients + loss
__global__ void __launch_bounds__(512) k_coef(const int* __restrict__ labels, const int* __restrict__ cams,
                                              const int* __restrict__ epoch, const int* __restrict__ lr,
                                              const unsigned long long* __restrict__ rowbest,
                                              const unsigned long long* __restrict__ colbest,
                                              const float* __restrict__ logits, const float* __restrict__ logZ,
                                              float* __restrict__ alpha, int* __restrict__ idxs,
                                              float* __restrict__ coefA, int* __restrict__ idxc,
                                              float* __restrict__ out0) {
  int b = threadIdx.x;
  float part = 0.f;
  if (epoch[0] > WARM) {
    int cam = cams[b] - 1, lab = labels[b] - 1;
    float lrf = (float)lr[0];
    float ce_self = logZ[b * NC + cam] - logits[(size_t)b * LW + cam * NPAD + lab];
    alpha[b] = lrf * (1.f - ce_self);
    idxs[b]  = cam * NID + lab;
    part = ce_self;
    for (int c = 0; c < NC; ++c) {
      unsigned long long pk = rowbest[(size_t)(cam * NC + c) * NPAD + lab];
      int   rb = unpackIdx(pk);
      float sb = unpackVal(pk);
      unsigned long long ck = colbest[(size_t)(cam * NC + c) * NPAD + rb];
      bool valid = (unpackIdx(ck) == lab) && (sb > 0.5f) && (c != cam);
      float ce_cr = logZ[b * NC + c] - logits[(size_t)b * LW + c * NPAD + rb];
      float w = valid ? sb : 0.f;
      part += w * ce_cr;
      coefA[b * NC + c] = w * lrf * (1.f - ce_cr);
      idxc[b * NC + c]  = c * NID + rb;
    }
  } else {
    alpha[b] = 0.f; idxs[b] = 0;
    for (int c = 0; c < NC; ++c) { coefA[b * NC + c] = 0.f; idxc[b * NC + c] = 0; }
  }
  __shared__ float red[512];
  red[b] = part; __syncthreads();
  for (int s = 256; s > 0; s >>= 1) { if (b < s) red[b] += red[b + s]; __syncthreads(); }
  if (b == 0) out0[0] = red[0] / (float)NB;
}

// deterministic scatter: each thread owns one feature column d, loops batch serially
__global__ void k_scatter(const float* __restrict__ fnorm,
                          const float* __restrict__ alpha, const int* __restrict__ idxs,
                          const float* __restrict__ coefA, const int* __restrict__ idxc,
                          const int* __restrict__ epoch, float* __restrict__ x) {
  if (epoch[0] <= WARM) return;
  int d = blockIdx.x * 256 + threadIdx.x;    // 0..DIM-1
  for (int b = 0; b < NB; ++b) {
    float fv = fnorm[(size_t)b * DIM + d];
    x[(size_t)idxs[b] * DIM + d] -= alpha[b] * fv;
#pragma unroll
    for (int c = 0; c < NC; ++c) {
      float w = coefA[b * NC + c];
      if (w != 0.f) x[(size_t)idxc[b * NC + c] * DIM + d] -= w * fv;
    }
  }
}

// ---------------- launcher ----------------
extern "C" void kernel_launch(void* const* d_in, const int* in_sizes, int n_in,
                              void* d_out, int out_size, void* d_ws, size_t ws_size,
                              hipStream_t stream) {
  const float* features = (const float*)d_in[0];
  const int*   labels   = (const int*)d_in[1];
  const int*   cams     = (const int*)d_in[2];
  const float* intra    = (const float*)d_in[3];
  const float* cross    = (const float*)d_in[4];
  const int*   epoch    = (const int*)d_in[5];
  const int*   lr       = (const int*)d_in[6];
  float* out = (float*)d_out;

  char* w = (char*)d_ws;
  float*  fnorm = (float*)w;              w += (size_t)NB * DIM * 4;
  __bf16* fbf   = (__bf16*)w;             w += (size_t)NB * DIM * 2;
  __bf16* Abf   = (__bf16*)w;             w += (size_t)NC * NPAD * DIM * 2;
  unsigned long long* rowbest = (unsigned long long*)w; w += (size_t)NPAIR * NPAD * 8;
  unsigned long long* colbest = (unsigned long long*)w; w += (size_t)NPAIR * NPAD * 8;
  float* logitsb = (float*)w;             w += (size_t)NB * LW * 4;
  float* logZ    = (float*)w;             w += (size_t)NB * NC * 4;
  float* alpha   = (float*)w;             w += (size_t)NB * 4;
  int*   idxs    = (int*)w;               w += (size_t)NB * 4;
  float* coefA   = (float*)w;             w += (size_t)NB * NC * 4;
  int*   idxc    = (int*)w;               w += (size_t)NB * NC * 4;

  int nz = NPAIR * NPAD * 2;               // rowbest + colbest are contiguous
  k_zero64<<<(nz + 255) / 256, 256, 0, stream>>>(rowbest, nz);
  k_norm_feat<<<NB, 256, 0, stream>>>(features, fnorm, fbf);
  k_anchors_bf16<<<NC * NPAD, 256, 0, stream>>>(cross, Abf);
  k_write_x<<<NC * NID, 256, 0, stream>>>(cross, intra, epoch, out + 1);

  dim3 gS(NPAIR, NPAD / 64);
  k_pairsim<<<gS, 256, 0, stream>>>(Abf, rowbest, colbest);

  dim3 gL(NC, NB / 64);
  k_logits<<<gL, 256, 0, stream>>>(fbf, Abf, logitsb);

  k_lse<<<NB * NC, 128, 0, stream>>>(logitsb, logZ);
  k_coef<<<1, 512, 0, stream>>>(labels, cams, epoch, lr, rowbest, colbest,
                                logitsb, logZ, alpha, idxs, coefA, idxc, out);
  k_scatter<<<DIM / 256, 256, 0, stream>>>(fnorm, alpha, idxs, coefA, idxc, epoch, out + 1);
}
